// SurrogateDistillClassifier_27625229648168
// MI455X (gfx1250) — compile-verified, matrix operand order checked by reading, not run on hardware
//
#include <hip/hip_runtime.h>
#include <hip/hip_bf16.h>
#include <stdint.h>

// ---------------- problem constants ----------------
#define N_ROWS 32768
#define DIM    768
#define CLS    1024
#define K_MU   0.9f
#define K_LAM  0.5f

typedef __attribute__((ext_vector_type(16))) __bf16 v16bf;
typedef __attribute__((ext_vector_type(8)))  float  v8f;

union FragU { v16bf v; uint4 q[2]; };

__device__ __forceinline__ unsigned short f2bf(float f) {
  unsigned u = __float_as_uint(f);
  unsigned r = u + 0x7FFFu + ((u >> 16) & 1u);   // round-to-nearest-even
  return (unsigned short)(r >> 16);
}

// ---------------- workspace layout (bytes, all 256-aligned) ----------------
constexpr size_t SZ_SUMS  = (size_t)CLS * DIM * 4;      // seg sums (row C-1 unused pad)
constexpr size_t SZ_CNTS  = 4096;                       // per-class float counts
constexpr size_t SZ_ACC   = 256;                        // loss accumulators
constexpr size_t SZ_VT    = (size_t)DIM * CLS * 2;      // v_pad^T bf16 [D][C]
constexpr size_t OFF_SUMS = 0;
constexpr size_t OFF_CNTS = OFF_SUMS + SZ_SUMS;
constexpr size_t OFF_ACC  = OFF_CNTS + SZ_CNTS;
constexpr size_t OFF_VT   = OFF_ACC + SZ_ACC;
constexpr size_t ZERO_BYTES = OFF_VT + SZ_VT;           // contiguous zero region
constexpr size_t OFF_XB   = ZERO_BYTES;                               // x bf16
constexpr size_t OFF_XCB  = OFF_XB  + (size_t)N_ROWS * DIM * 2;       // x_ctx bf16
constexpr size_t OFF_HB   = OFF_XCB + (size_t)N_ROWS * DIM * 2;       // h bf16
constexpr size_t OFF_MIXB = OFF_HB  + (size_t)N_ROWS * DIM * 2;       // mixed bf16
constexpr size_t OFF_WB   = OFF_MIXB+ (size_t)N_ROWS * DIM * 2;       // softmax(w) bf16 [N,C]
constexpr size_t OFF_CTXL = OFF_WB  + (size_t)N_ROWS * CLS * 2;       // ctx_logits f32
constexpr size_t OFF_WLOG = OFF_CTXL+ (size_t)N_ROWS * CLS * 4;       // w_logits f32
constexpr size_t OFF_WCTXB= OFF_WLOG+ (size_t)N_ROWS * CLS * 4;       // W_ctx bf16
constexpr size_t OFF_W1B  = OFF_WCTXB + (size_t)CLS * DIM * 2;        // W1 bf16
constexpr size_t OFF_W2B  = OFF_W1B   + (size_t)DIM * DIM * 2;        // W2 bf16
constexpr size_t OFF_CLSB = OFF_W2B   + (size_t)CLS * DIM * 2;        // classifier bf16
constexpr size_t OFF_G    = OFF_CLSB  + (size_t)CLS * DIM * 2;        // g [N] f32

// ---------------- utility kernels ----------------
__global__ void k_zero_u32(uint32_t* p, int n) {
  int i = blockIdx.x * 256 + threadIdx.x;
  if (i < n) p[i] = 0u;
}

__global__ void k_f32_to_bf16(const float* __restrict__ s, unsigned short* __restrict__ d, int n) {
  int i = blockIdx.x * 256 + threadIdx.x;
  if (i < n) d[i] = f2bf(s[i]);
}

// segment sums + counts (classes 1..C-1 -> rows 0..C-2; label 0 dropped)
__global__ void k_seg_sum(const float* __restrict__ x, const int* __restrict__ labels,
                          float* __restrict__ sums, float* __restrict__ cnts) {
  int i = blockIdx.x * 256 + threadIdx.x;
  if (i >= N_ROWS * DIM) return;
  int n = i / DIM, d = i - n * DIM;
  int lab = labels[n];
  if (lab >= 1) {
    atomicAdd(&sums[(size_t)(lab - 1) * DIM + d], x[i]);
    if (d == 0) atomicAdd(&cnts[lab - 1], 1.0f);
  }
}

// fused: x -> bf16  and  g = 0.5*sigmoid(x@W_g + b_g)    (one wave per row)
__global__ void __launch_bounds__(256) k_gate_conv(const float* __restrict__ x,
                                                   const float* __restrict__ Wg,
                                                   const float* __restrict__ bg,
                                                   unsigned short* __restrict__ xb,
                                                   float* __restrict__ g) {
  int wid = threadIdx.x >> 5, lane = threadIdx.x & 31;
  int row = blockIdx.x * 8 + wid;
  const float* xr = x + (size_t)row * DIM;
  unsigned short* xbr = xb + (size_t)row * DIM;
  float s = 0.f;
  for (int d = lane; d < DIM; d += 32) {
    float v = xr[d];
    xbr[d] = f2bf(v);
    s += v * Wg[d];
  }
  for (int m = 16; m; m >>= 1) s += __shfl_xor(s, m, 32);
  if (lane == 0) g[row] = 0.5f / (1.f + expf(-(s + bg[0])));
}

// history EMA -> v, stored transposed+padded: vT[d*C + 0] = 0 (pre-zeroed), vT[d*C + c] = v[c-1][d]
__global__ void k_build_vt(const float* __restrict__ sums, const float* __restrict__ cnts,
                           const float* __restrict__ hist, const int* __restrict__ hcnt,
                           unsigned short* __restrict__ vT) {
  int i = blockIdx.x * 256 + threadIdx.x;
  if (i >= (CLS - 1) * DIM) return;
  int r = i / DIM, d = i - r * DIM;
  float cnt = cnts[r];
  float hv  = hist[i];
  int   nc  = hcnt[r];
  float nh;
  if (cnt > 0.f) { nh = K_MU * hv + sums[i] / cnt; nc += 1; }
  else           { nh = hv; }
  if (nc < 1) nc = 1;
  float denom = 1.f - __powf(K_MU, (float)nc);
  float vv = nh * (1.f - K_MU) / denom;
  vT[(size_t)d * CLS + (r + 1)] = f2bf(vv);
}

// ---------------- WMMA bf16 GEMM: out[N,Cout] = A[N,K] @ Bt[Cout,K]^T (+bias, epilogue) ----------
// wave tile 32(M) x 64(N): 2 A-fragments, 4 B-fragments, 8 accumulators
//   -> 12 b128 loads per 8 WMMAs (1.5 loads/WMMA, 21.3 FLOP/byte at the VMEM port).
// block = 8 waves arranged 2(M) x 4(N) -> 64 x 256 block tile.
// EPI: 0 = f32 + bias; 1 = gelu(f32+bias) -> bf16; 2 = g*x + (1-g)*acc -> bf16; 3 = f32 plain.
template <int EPI>
__global__ void __launch_bounds__(256) k_wmma_gemm(const unsigned short* __restrict__ A,
                                                   const unsigned short* __restrict__ Bt,
                                                   const float* __restrict__ bias,
                                                   float* __restrict__ outF,
                                                   unsigned short* __restrict__ outB,
                                                   const float* __restrict__ xsrc,
                                                   const float* __restrict__ g,
                                                   int K, int Cout) {
  int lane = threadIdx.x & 31;
  int wid  = threadIdx.x >> 5;
  int l  = lane & 15;
  int hf = lane >> 4;
  int rowBase = blockIdx.x * 64  + (wid & 1) * 32;   // 2 M-waves
  int colBase = blockIdx.y * 256 + (wid >> 1) * 64;  // 4 N-waves

  const unsigned short* arow0 = A  + (size_t)(rowBase + l) * K;
  const unsigned short* arow1 = arow0 + (size_t)16 * K;
  const unsigned short* brow0 = Bt + (size_t)(colBase + l) * K;
  const unsigned short* brow1 = brow0 + (size_t)16 * K;
  const unsigned short* brow2 = brow0 + (size_t)32 * K;
  const unsigned short* brow3 = brow0 + (size_t)48 * K;

  v8f acc[2][4];
#pragma unroll
  for (int i = 0; i < 2; ++i)
#pragma unroll
    for (int j = 0; j < 4; ++j)
      acc[i][j] = (v8f){0.f, 0.f, 0.f, 0.f, 0.f, 0.f, 0.f, 0.f};

  for (int k0 = 0; k0 < K; k0 += 32) {
    int base = k0 + hf * 8;          // lane's K-chunk start (ISA 16-bit A/B fragment layout)
    __builtin_prefetch(arow0 + base + 64, 0, 3);
    __builtin_prefetch(arow1 + base + 64, 0, 3);
    FragU a0, a1, b0, b1, b2, b3;
    a0.q[0] = *(const uint4*)(arow0 + base); a0.q[1] = *(const uint4*)(arow0 + base + 16);
    a1.q[0] = *(const uint4*)(arow1 + base); a1.q[1] = *(const uint4*)(arow1 + base + 16);
    b0.q[0] = *(const uint4*)(brow0 + base); b0.q[1] = *(const uint4*)(brow0 + base + 16);
    b1.q[0] = *(const uint4*)(brow1 + base); b1.q[1] = *(const uint4*)(brow1 + base + 16);
    b2.q[0] = *(const uint4*)(brow2 + base); b2.q[1] = *(const uint4*)(brow2 + base + 16);
    b3.q[0] = *(const uint4*)(brow3 + base); b3.q[1] = *(const uint4*)(brow3 + base + 16);
    acc[0][0] = __builtin_amdgcn_wmma_f32_16x16x32_bf16(false, a0.v, false, b0.v, (short)0, acc[0][0], false, false);
    acc[0][1] = __builtin_amdgcn_wmma_f32_16x16x32_bf16(false, a0.v, false, b1.v, (short)0, acc[0][1], false, false);
    acc[0][2] = __builtin_amdgcn_wmma_f32_16x16x32_bf16(false, a0.v, false, b2.v, (short)0, acc[0][2], false, false);
    acc[0][3] = __builtin_amdgcn_wmma_f32_16x16x32_bf16(false, a0.v, false, b3.v, (short)0, acc[0][3], false, false);
    acc[1][0] = __builtin_amdgcn_wmma_f32_16x16x32_bf16(false, a1.v, false, b0.v, (short)0, acc[1][0], false, false);
    acc[1][1] = __builtin_amdgcn_wmma_f32_16x16x32_bf16(false, a1.v, false, b1.v, (short)0, acc[1][1], false, false);
    acc[1][2] = __builtin_amdgcn_wmma_f32_16x16x32_bf16(false, a1.v, false, b2.v, (short)0, acc[1][2], false, false);
    acc[1][3] = __builtin_amdgcn_wmma_f32_16x16x32_bf16(false, a1.v, false, b3.v, (short)0, acc[1][3], false, false);
  }

  // C/D layout: VGPR r -> row = rowBase + i*16 + r + 8*hf, col = colBase + j*16 + l
#pragma unroll
  for (int i = 0; i < 2; ++i) {
#pragma unroll
    for (int j = 0; j < 4; ++j) {
      int ocol = colBase + j * 16 + l;
#pragma unroll
      for (int r = 0; r < 8; ++r) {
        int orow = rowBase + i * 16 + r + 8 * hf;
        float val = acc[i][j][r];
        if (EPI == 0) {
          outF[(size_t)orow * Cout + ocol] = val + bias[ocol];
        } else if (EPI == 1) {
          float t = val + bias[ocol];
          float ge = 0.5f * t * (1.f + erff(t * 0.70710678118654752f));
          outB[(size_t)orow * Cout + ocol] = f2bf(ge);
        } else if (EPI == 2) {
          float gg = g[orow];
          float xv = xsrc[(size_t)orow * Cout + ocol];
          outB[(size_t)orow * Cout + ocol] = f2bf(gg * xv + (1.f - gg) * val);
        } else {
          outF[(size_t)orow * Cout + ocol] = val;
        }
      }
    }
  }
}

// ---------------- row reductions: softmax(w_logits)->bf16, CE(ctx), MSE, valid ----------------
__global__ void __launch_bounds__(256) k_row_stats(const float* __restrict__ ctx,
                                                   const float* __restrict__ wl,
                                                   const int* __restrict__ labels,
                                                   unsigned short* __restrict__ wb,
                                                   float* __restrict__ acc) {
  int wid = threadIdx.x >> 5, lane = threadIdx.x & 31;
  int row = blockIdx.x * 8 + wid;
  const float* cr = ctx + (size_t)row * CLS;
  const float* wr = wl  + (size_t)row * CLS;
  float mw = -3.4e38f, mc = -3.4e38f, sqd = 0.f;
  for (int c = lane; c < CLS; c += 32) {
    float a = wr[c], b = cr[c];
    mw = fmaxf(mw, a); mc = fmaxf(mc, b);
    float d = b - a; sqd += d * d;
  }
  for (int m = 16; m; m >>= 1) {
    mw  = fmaxf(mw, __shfl_xor(mw, m, 32));
    mc  = fmaxf(mc, __shfl_xor(mc, m, 32));
    sqd += __shfl_xor(sqd, m, 32);
  }
  float sw = 0.f, sc = 0.f;
  for (int c = lane; c < CLS; c += 32) { sw += expf(wr[c] - mw); sc += expf(cr[c] - mc); }
  for (int m = 16; m; m >>= 1) { sw += __shfl_xor(sw, m, 32); sc += __shfl_xor(sc, m, 32); }
  float inv = 1.f / sw;
  unsigned short* wbr = wb + (size_t)row * CLS;
  for (int c = lane; c < CLS; c += 32) wbr[c] = f2bf(expf(wr[c] - mw) * inv);
  if (lane == 0) {
    int lab = labels[row];
    if (lab >= 0) {
      float nll = (mc + logf(sc)) - cr[lab];
      atomicAdd(&acc[1], nll);   // CE(ctx_logits)
      atomicAdd(&acc[2], sqd);   // MSE numerator
      atomicAdd(&acc[3], 1.f);   // valid count
    }
  }
}

__global__ void __launch_bounds__(256) k_ce_rows(const float* __restrict__ lgt,
                                                 const int* __restrict__ labels,
                                                 float* __restrict__ acc) {
  int wid = threadIdx.x >> 5, lane = threadIdx.x & 31;
  int row = blockIdx.x * 8 + wid;
  const float* lr = lgt + (size_t)row * CLS;
  float mx = -3.4e38f;
  for (int c = lane; c < CLS; c += 32) mx = fmaxf(mx, lr[c]);
  for (int m = 16; m; m >>= 1) mx = fmaxf(mx, __shfl_xor(mx, m, 32));
  float s = 0.f;
  for (int c = lane; c < CLS; c += 32) s += expf(lr[c] - mx);
  for (int m = 16; m; m >>= 1) s += __shfl_xor(s, m, 32);
  if (lane == 0) {
    int lab = labels[row];
    if (lab >= 0) atomicAdd(&acc[0], (mx + logf(s)) - lr[lab]);
  }
}

__global__ void k_finalize(const float* __restrict__ acc, float* __restrict__ out) {
  if (threadIdx.x == 0 && blockIdx.x == 0) {
    float nv = fmaxf(acc[3], 1.f);
    out[0] = acc[0] / nv + acc[1] / nv + K_LAM * (acc[2] / (nv * (float)CLS));
  }
}

// ---------------- launcher ----------------
extern "C" void kernel_launch(void* const* d_in, const int* in_sizes, int n_in,
                              void* d_out, int out_size, void* d_ws, size_t ws_size,
                              hipStream_t stream) {
  const float* x          = (const float*)d_in[0];
  const float* x_ctx      = (const float*)d_in[1];
  const int*   labels     = (const int*)  d_in[2];
  const float* classifier = (const float*)d_in[3];
  const float* W_ctx      = (const float*)d_in[4];
  const float* b_ctx      = (const float*)d_in[5];
  const float* W_g        = (const float*)d_in[6];
  const float* b_g        = (const float*)d_in[7];
  const float* W1         = (const float*)d_in[8];
  const float* b1         = (const float*)d_in[9];
  const float* W2         = (const float*)d_in[10];
  const float* b2         = (const float*)d_in[11];
  const float* hist       = (const float*)d_in[12];
  const int*   hcnt       = (const int*)  d_in[13];
  (void)in_sizes; (void)n_in; (void)out_size; (void)ws_size;

  char* ws = (char*)d_ws;
  float*          sums  = (float*)(ws + OFF_SUMS);
  float*          cnts  = (float*)(ws + OFF_CNTS);
  float*          accp  = (float*)(ws + OFF_ACC);
  unsigned short* vT    = (unsigned short*)(ws + OFF_VT);
  unsigned short* xb    = (unsigned short*)(ws + OFF_XB);
  unsigned short* xcb   = (unsigned short*)(ws + OFF_XCB);
  unsigned short* hb    = (unsigned short*)(ws + OFF_HB);
  unsigned short* mixb  = (unsigned short*)(ws + OFF_MIXB);
  unsigned short* wb    = (unsigned short*)(ws + OFF_WB);
  float*          ctxl  = (float*)(ws + OFF_CTXL);
  float*          wlog  = (float*)(ws + OFF_WLOG);
  unsigned short* wctxb = (unsigned short*)(ws + OFF_WCTXB);
  unsigned short* w1b   = (unsigned short*)(ws + OFF_W1B);
  unsigned short* w2b   = (unsigned short*)(ws + OFF_W2B);
  unsigned short* clsb  = (unsigned short*)(ws + OFF_CLSB);
  float*          gbuf  = (float*)(ws + OFF_G);

  float* out    = (float*)d_out;
  float* logits = out + 1;   // (loss, logits) concatenated

  const int ND = N_ROWS * DIM;
  const int zero_words = (int)(ZERO_BYTES / 4);

  k_zero_u32<<<(zero_words + 255) / 256, 256, 0, stream>>>((uint32_t*)ws, zero_words);
  k_seg_sum<<<(ND + 255) / 256, 256, 0, stream>>>(x, labels, sums, cnts);
  k_gate_conv<<<N_ROWS / 8, 256, 0, stream>>>(x, W_g, b_g, xb, gbuf);
  k_f32_to_bf16<<<(ND + 255) / 256, 256, 0, stream>>>(x_ctx, xcb, ND);
  k_f32_to_bf16<<<(CLS * DIM + 255) / 256, 256, 0, stream>>>(W_ctx, wctxb, CLS * DIM);
  k_f32_to_bf16<<<(DIM * DIM + 255) / 256, 256, 0, stream>>>(W1, w1b, DIM * DIM);
  k_f32_to_bf16<<<(CLS * DIM + 255) / 256, 256, 0, stream>>>(W2, w2b, CLS * DIM);
  k_f32_to_bf16<<<(CLS * DIM + 255) / 256, 256, 0, stream>>>(classifier, clsb, CLS * DIM);
  k_build_vt<<<((CLS - 1) * DIM + 255) / 256, 256, 0, stream>>>(sums, cnts, hist, hcnt, vT);

  dim3 blk(256);
  // ctx_logits = x_ctx @ W_ctx^T + b_ctx
  k_wmma_gemm<0><<<dim3(N_ROWS / 64, CLS / 256), blk, 0, stream>>>(xcb, wctxb, b_ctx, ctxl, nullptr, nullptr, nullptr, DIM, CLS);
  // h = gelu(x @ W1^T + b1) -> bf16
  k_wmma_gemm<1><<<dim3(N_ROWS / 64, DIM / 256), blk, 0, stream>>>(xb, w1b, b1, nullptr, hb, nullptr, nullptr, DIM, DIM);
  // w_logits = h @ W2^T + b2
  k_wmma_gemm<0><<<dim3(N_ROWS / 64, CLS / 256), blk, 0, stream>>>(hb, w2b, b2, wlog, nullptr, nullptr, nullptr, DIM, CLS);
  // softmax(w_logits)->wb ; CE(ctx) ; MSE ; valid-count
  k_row_stats<<<N_ROWS / 8, 256, 0, stream>>>(ctxl, wlog, labels, wb, accp);
  // mixed = g*x + (1-g)*(w @ v_pad) -> bf16   (K = C, Cout = D, B = vT)
  k_wmma_gemm<2><<<dim3(N_ROWS / 64, DIM / 256), blk, 0, stream>>>(wb, vT, nullptr, nullptr, mixb, x, gbuf, CLS, DIM);
  // logits = mixed @ classifier^T  (straight into d_out+1)
  k_wmma_gemm<3><<<dim3(N_ROWS / 64, CLS / 256), blk, 0, stream>>>(mixb, clsb, nullptr, logits, nullptr, nullptr, nullptr, DIM, CLS);
  k_ce_rows<<<N_ROWS / 8, 256, 0, stream>>>(logits, labels, accp);
  k_finalize<<<1, 1, 0, stream>>>(accp, out);
}
